// KernalAttention_35424890257839
// MI455X (gfx1250) — matrix operand order repaired and checked by reading, not compiled, not run on hardware
//
#include <hip/hip_runtime.h>
#include <hip/hip_bf16.h>

typedef __attribute__((ext_vector_type(16))) _Float16 v16h;
typedef __attribute__((ext_vector_type(8)))  _Float16 v8h;
typedef __attribute__((ext_vector_type(8)))  float    v8f;
typedef __attribute__((ext_vector_type(4)))  float    v4f;

#define R_FEAT 256
#define D_DIM  64
#define DV_DIM 64
#define CHUNK  64
#define LDT    80    // stride (halfs) of transposed 64x64 operand stages: 160B, 16B-aligned, bank-spread
#define LDP1   80    // stride (halfs) of P[r][col] in kernel 1
#define LDP2   264   // stride (halfs) of PT[l][r] in kernel 2: 528B, 16B-aligned, bank-spread

// phi(x) = Dn*exp(A + B*(w.x) - ||x||^2), A=0.1, B=sqrt(0.6), Dn=0.6^16; C0=Dn*exp(A)
#define BC_CONST 0.7745966692414834f
#define C0_CONST (2.8211099074244517e-4f * 1.1051709180756477f)

#define WMMA(A, B, C) \
  __builtin_amdgcn_wmma_f32_16x16x32_f16(false, (A), false, (B), (short)0, (C), false, false)

__device__ __forceinline__ v16h frag_cat(v8h lo, v8h hi) {
  return __builtin_shufflevector(lo, hi, 0, 1, 2, 3, 4, 5, 6, 7, 8, 9, 10, 11, 12, 13, 14, 15);
}

// A-operand 16x32: A[m][k]; storage row-major over k, stride ld. 2x ds/global b128.
__device__ __forceinline__ v16h load_a_f16(const _Float16* __restrict__ base, int ld, int koff) {
  const int lane = threadIdx.x & 31;
  const _Float16* p = base + (lane & 15) * ld + koff + ((lane & 16) ? 16 : 0);
  return frag_cat(*(const v8h*)p, *(const v8h*)(p + 8));
}

// B-operand 32x16: B[k][n]; storage with column n contiguous over k, stride ld.
__device__ __forceinline__ v16h load_b_f16(const _Float16* __restrict__ base, int ld, int koff) {
  const int lane = threadIdx.x & 31;
  const _Float16* p = base + (lane & 15) * ld + koff + ((lane & 16) ? 16 : 0);
  return frag_cat(*(const v8h*)p, *(const v8h*)(p + 8));
}

__device__ __forceinline__ v8h cvt8h(const float* __restrict__ p) {
  v4f x = *(const v4f*)p;
  v4f y = *(const v4f*)(p + 4);
  v8h r;
  r[0] = (_Float16)x[0]; r[1] = (_Float16)x[1]; r[2] = (_Float16)x[2]; r[3] = (_Float16)x[3];
  r[4] = (_Float16)y[0]; r[5] = (_Float16)y[1]; r[6] = (_Float16)y[2]; r[7] = (_Float16)y[3];
  return r;
}

__device__ __forceinline__ v16h load_a_f32(const float* __restrict__ base, int ld, int koff) {
  const int lane = threadIdx.x & 31;
  const float* p = base + (lane & 15) * ld + koff + ((lane & 16) ? 16 : 0);
  return frag_cat(cvt8h(p), cvt8h(p + 8));
}

__device__ __forceinline__ v16h load_b_f32(const float* __restrict__ base, int ld, int koff) {
  const int lane = threadIdx.x & 31;
  const float* p = base + (lane & 15) * ld + koff + ((lane & 16) ? 16 : 0);
  return frag_cat(cvt8h(p), cvt8h(p + 8));
}

// ---- Kernel 0: zero the kvT accumulator ------------------------------------
__global__ void favor_zero_kv(float* __restrict__ kvT) {
  const int i = blockIdx.x * blockDim.x + threadIdx.x;
  if (i < R_FEAT * DV_DIM) kvT[i] = 0.0f;
}

// ---- Kernel 1: kvT[64 dv][256 r] = (phiK @ V)^T ----------------------------
__global__ __launch_bounds__(256)
void favor_kv_kernel(const float* __restrict__ K, const float* __restrict__ V,
                     const float* __restrict__ W, float* __restrict__ kvT,
                     int L, int chunksPerWG) {
  __shared__ _Float16 KldsT[CHUNK * LDT];   // [col][d]
  __shared__ _Float16 VldsT[DV_DIM * LDT];  // [dv][row]
  __shared__ _Float16 P[R_FEAT * LDP1];     // phiK chunk [r][col]
  __shared__ float    sqv[CHUNK];

  const int tid  = threadIdx.x;
  const int wave = tid >> 5;
  const int lane = tid & 31;
  const int nidx = lane & 15;
  const int hi8  = (lane & 16) ? 8 : 0;

  // Hoist W^T B-fragments (constant over all chunks): b[j] = W[(nt*16+n)*64 + k]
  v16h wfrag[2][2];
#pragma unroll
  for (int nn = 0; nn < 2; ++nn)
#pragma unroll
    for (int ks = 0; ks < 2; ++ks)
      wfrag[nn][ks] = load_b_f32(W + (wave * 2 + nn) * 16 * D_DIM, D_DIM, ks * 32);

  v8f kvacc[8];
#pragma unroll
  for (int t = 0; t < 8; ++t) kvacc[t] = (v8f){};

  for (int ci = 0; ci < chunksPerWG; ++ci) {
    const int col0 = (blockIdx.x * chunksPerWG + ci) * CHUNK;

    if (ci + 1 < chunksPerWG) {  // gfx1250 global_prefetch_b8 for next chunk
      __builtin_prefetch(&K[(size_t)(tid & 63) * L + col0 + CHUNK], 0, 3);
      __builtin_prefetch(&V[(size_t)(col0 + CHUNK + (tid & 63)) * DV_DIM], 0, 3);
    }

    // Stage transposed: contiguous global reads, K-contiguous LDS rows.
    for (int i = tid; i < D_DIM * CHUNK; i += 256) {
      const int d = i >> 6, c = i & 63;
      KldsT[c * LDT + d] = (_Float16)K[(size_t)d * L + col0 + c];
    }
    for (int i = tid; i < CHUNK * DV_DIM; i += 256) {
      const int r = i >> 6, dv = i & 63;
      VldsT[dv * LDT + r] = (_Float16)V[(size_t)(col0 + r) * DV_DIM + dv];
    }
    __syncthreads();

    if (tid < CHUNK) {  // ||k||^2 per column from staged row (contiguous LDS reads)
      float s = 0.0f;
      for (int d = 0; d < D_DIM; ++d) {
        const float x = (float)KldsT[tid * LDT + d];
        s += x * x;
      }
      sqv[tid] = s;
    }
    __syncthreads();

    v8f sqreg[4];
#pragma unroll
    for (int mt = 0; mt < 4; ++mt) sqreg[mt] = *(const v8f*)&sqv[mt * 16 + hi8];

    // GEMM1 (S^T orientation): ST[col][r] = Kc^T @ W^T, fused phi -> P[r][col]
#pragma unroll
    for (int nn = 0; nn < 2; ++nn) {
      const int r = (wave * 2 + nn) * 16 + nidx;
#pragma unroll
      for (int mt = 0; mt < 4; ++mt) {
        v8f acc = {};
        acc = WMMA(load_a_f16(KldsT + mt * 16 * LDT, LDT, 0),  wfrag[nn][0], acc);
        acc = WMMA(load_a_f16(KldsT + mt * 16 * LDT, LDT, 32), wfrag[nn][1], acc);
        v8h ph;
#pragma unroll
        for (int v = 0; v < 8; ++v)
          ph[v] = (_Float16)(C0_CONST * __expf(BC_CONST * acc[v] - sqreg[mt][v]));
        *(v8h*)&P[r * LDP1 + mt * 16 + hi8] = ph;  // one ds_store_b128
      }
    }
    __syncthreads();

    // GEMM2: kvT[dv][r] += V^T @ P-rows-as-B
#pragma unroll
    for (int nn = 0; nn < 2; ++nn) {
      const int nt = wave * 2 + nn;
      const v16h b0 = load_b_f16(P + nt * 16 * LDP1, LDP1, 0);
      const v16h b1 = load_b_f16(P + nt * 16 * LDP1, LDP1, 32);
#pragma unroll
      for (int mt = 0; mt < 4; ++mt) {
        const int t = nn * 4 + mt;
        kvacc[t] = WMMA(load_a_f16(VldsT + mt * 16 * LDT, LDT, 0),  b0, kvacc[t]);
        kvacc[t] = WMMA(load_a_f16(VldsT + mt * 16 * LDT, LDT, 32), b1, kvacc[t]);
      }
    }
    __syncthreads();
  }

#pragma unroll
  for (int nn = 0; nn < 2; ++nn) {
    const int r = (wave * 2 + nn) * 16 + nidx;
#pragma unroll
    for (int mt = 0; mt < 4; ++mt)
#pragma unroll
      for (int v = 0; v < 8; ++v)
        atomicAdd(&kvT[(mt * 16 + hi8 + v) * R_FEAT + r], kvacc[nn * 4 + mt][v]);
  }
}

// ---- Kernel 2: out[L][64] = phiQ^T @ kv  (computed as out^T tiles) ---------
__global__ __launch_bounds__(256)
void favor_out_kernel(const float* __restrict__ Q, const float* __restrict__ W,
                      const float* __restrict__ kvT, float* __restrict__ out,
                      int L) {
  __shared__ _Float16 QldsT[CHUNK * LDT];   // [l][d]
  __shared__ _Float16 PT[CHUNK * LDP2];     // phiQ chunk transposed: [l][r]
  __shared__ float    sqv[CHUNK];

  const int tid  = threadIdx.x;
  const int wave = tid >> 5;
  const int lane = tid & 31;
  const int nidx = lane & 15;
  const int hi8  = (lane & 16) ? 8 : 0;
  const int col0 = blockIdx.x * CHUNK;      // Q columns == output rows

  for (int i = tid; i < D_DIM * CHUNK; i += 256) {
    const int d = i >> 6, c = i & 63;
    QldsT[c * LDT + d] = (_Float16)Q[(size_t)d * L + col0 + c];
  }
  __syncthreads();

  if (tid < CHUNK) {
    float s = 0.0f;
    for (int d = 0; d < D_DIM; ++d) {
      const float x = (float)QldsT[tid * LDT + d];
      s += x * x;
    }
    sqv[tid] = s;
  }
  __syncthreads();

  // GEMM1 (S orientation): S[r][l] = W @ Qc, fused phi -> PT[l][r]
  // Wave covers mt (r-tiles) in {wave*2, wave*2+1}; nt (l-tiles) 0..3.
  v16h wa[2][2];
#pragma unroll
  for (int mm = 0; mm < 2; ++mm)
#pragma unroll
    for (int ks = 0; ks < 2; ++ks)
      wa[mm][ks] = load_a_f32(W + (wave * 2 + mm) * 16 * D_DIM, D_DIM, ks * 32);

#pragma unroll
  for (int mm = 0; mm < 2; ++mm) {
    const int mt = wave * 2 + mm;
#pragma unroll
    for (int nt = 0; nt < 4; ++nt) {
      v8f acc = {};
      acc = WMMA(wa[mm][0], load_b_f16(QldsT + nt * 16 * LDT, LDT, 0),  acc);
      acc = WMMA(wa[mm][1], load_b_f16(QldsT + nt * 16 * LDT, LDT, 32), acc);
      const int l = nt * 16 + nidx;
      const float sq = sqv[l];
      v8h ph;
#pragma unroll
      for (int v = 0; v < 8; ++v)
        ph[v] = (_Float16)(C0_CONST * __expf(BC_CONST * acc[v] - sq));
      *(v8h*)&PT[l * LDP2 + mt * 16 + hi8] = ph;  // one ds_store_b128
    }
  }
  __syncthreads();

  // GEMM2 (out^T orientation): outT[dv][l] tiles = kvT(64x256) @ P(256x64)
  // A = kvT rows (global f32, contiguous), B = PT rows (contiguous).
#pragma unroll
  for (int t = 0; t < 2; ++t) {
    const int id = wave * 2 + t;
    const int mt = id >> 2;  // dv tile 0..3
    const int nt = id & 3;   // l  tile 0..3
    v8f acc = {};
#pragma unroll
    for (int ks = 0; ks < 8; ++ks)
      acc = WMMA(load_a_f32(kvT + mt * 16 * R_FEAT, R_FEAT, ks * 32),
                 load_b_f16(PT + nt * 16 * LDP2, LDP2, ks * 32), acc);
    // Lane holds row l fixed, dv contiguous over v -> packed 32B stores.
    const int l = nt * 16 + nidx;
    float* dst = out + (size_t)(col0 + l) * DV_DIM + mt * 16 + hi8;
    *(v4f*)dst       = (v4f){acc[0], acc[1], acc[2], acc[3]};
    *(v4f*)(dst + 4) = (v4f){acc[4], acc[5], acc[6], acc[7]};
  }
}

extern "C" void kernel_launch(void* const* d_in, const int* in_sizes, int n_in,
                              void* d_out, int out_size, void* d_ws, size_t ws_size,
                              hipStream_t stream) {
  const float* Q = (const float*)d_in[0];   // [64, L]
  const float* K = (const float*)d_in[1];   // [64, L]
  const float* V = (const float*)d_in[2];   // [L, 64]
  const float* W = (const float*)d_in[3];   // [256, 64]
  float* out = (float*)d_out;               // [L, 64]
  float* kvT = (float*)d_ws;                // [64 dv][256 r] accumulator

  const int L = in_sizes[0] / D_DIM;        // 65536
  const int chunks = L / CHUNK;             // 1024
  int grid1 = chunks < 128 ? chunks : 128;
  const int cpw = chunks / grid1;

  favor_zero_kv<<<(R_FEAT * DV_DIM + 255) / 256, 256, 0, stream>>>(kvT);
  favor_kv_kernel<<<grid1, 256, 0, stream>>>(K, V, W, kvT, L, cpw);
  favor_out_kernel<<<chunks, 256, 0, stream>>>(Q, W, kvT, out, L);
}
